// UGCGRUCell_90202903150932
// MI455X (gfx1250) — matrix operand order repaired and checked by reading, compiled, not run on hardware
//
#include <hip/hip_runtime.h>
#include <math.h>

// ---- problem constants (match reference) ----
#define NN   10000   // num nodes
#define UU   64      // hidden units
#define IND  2       // input dim
#define BB   16      // batch
#define FF   66      // IND + UU
#define KPAD 160     // F*M = 132 padded up to 5*32
#define KS   5       // K-steps of 32

typedef __attribute__((ext_vector_type(16))) _Float16 v16h;
typedef __attribute__((ext_vector_type(8)))  _Float16 v8h;
typedef __attribute__((ext_vector_type(8)))  float    v8f;

// ---- fast CDNA5 transcendentals (guarded; fallbacks always compile) ----
__device__ __forceinline__ float fast_rcp(float x) {
#if __has_builtin(__builtin_amdgcn_rcpf)
    return __builtin_amdgcn_rcpf(x);          // v_rcp_f32
#else
    return 1.f / x;
#endif
}
__device__ __forceinline__ float fast_sigmoid(float x) {
    return fast_rcp(1.f + __expf(-x));
}
__device__ __forceinline__ float fast_tanh(float x) {
#if __has_builtin(__builtin_amdgcn_tanhf)
    return __builtin_amdgcn_tanhf(x);         // v_tanh_f32 (CDNA5 trans op)
#else
    float e = __expf(2.f * x);
    return 1.f - 2.f * fast_rcp(e + 1.f);
#endif
}

// ---------------------------------------------------------------------------
// 1) Bayesian reparam + transpose weights to f16 [col][KPAD] (K-padded)
// ---------------------------------------------------------------------------
__global__ void reparam_kernel(const float* __restrict__ w1_mu, const float* __restrict__ w1_ls,
                               const float* __restrict__ e_w1,
                               const float* __restrict__ b1_mu, const float* __restrict__ b1_ls,
                               const float* __restrict__ e_b1,
                               const float* __restrict__ w2_mu, const float* __restrict__ w2_ls,
                               const float* __restrict__ e_w2,
                               const float* __restrict__ b2_mu, const float* __restrict__ b2_ls,
                               const float* __restrict__ e_b2,
                               _Float16* __restrict__ W1t, _Float16* __restrict__ W2t,
                               float* __restrict__ B1f, float* __restrict__ B2f)
{
    int idx = blockIdx.x * blockDim.x + threadIdx.x;
    const int n1 = 2 * UU * KPAD;   // 20480
    const int n2 = UU * KPAD;       // 10240
    if (idx < n1) {
        int j = idx / KPAD, k = idx % KPAD;
        float v = 0.f;
        if (k < FF * 2)
            v = w1_mu[k * 2 * UU + j] + __expf(w1_ls[k * 2 * UU + j]) * e_w1[k * 2 * UU + j];
        W1t[(size_t)j * KPAD + k] = (_Float16)v;
        return;
    }
    int i2 = idx - n1;
    if (i2 < n2) {
        int j = i2 / KPAD, k = i2 % KPAD;
        float v = 0.f;
        if (k < FF * 2)
            v = w2_mu[k * UU + j] + __expf(w2_ls[k * UU + j]) * e_w2[k * UU + j];
        W2t[(size_t)j * KPAD + k] = (_Float16)v;
        return;
    }
    int i3 = i2 - n2;
    if (i3 < 2 * UU) { B1f[i3] = b1_mu[i3] + __expf(b1_ls[i3]) * e_b1[i3]; return; }
    int i4 = i3 - 2 * UU;
    if (i4 < UU) { B2f[i4] = b2_mu[i4] + __expf(b2_ls[i4]) * e_b2[i4]; }
}

// ---------------------------------------------------------------------------
// 2) assemble x0a[n][f][b] = concat(inputs, hx) in f16; input rows also to x0b
// ---------------------------------------------------------------------------
__global__ void assemble_kernel(const float* __restrict__ inputs, const float* __restrict__ hx,
                                _Float16* __restrict__ x0a, _Float16* __restrict__ x0b)
{
    int idx = blockIdx.x * blockDim.x + threadIdx.x;
    const int total = NN * FF * BB;
    if (idx >= total) return;
    int b = idx % BB;
    int f = (idx / BB) % FF;
    int n = idx / (BB * FF);
    float v = (f < IND) ? inputs[(size_t)b * NN * IND + (size_t)n * IND + f]
                        : hx[(size_t)b * NN * UU + (size_t)n * UU + (f - IND)];
    _Float16 h = (_Float16)v;
    x0a[idx] = h;
    if (f < IND) x0b[idx] = h;
}

// ---------------------------------------------------------------------------
// 3) on-device CSR build: histogram -> single-block scan -> scatter
// ---------------------------------------------------------------------------
__global__ void csr_count_kernel(const int* __restrict__ erow, int* __restrict__ cnt, int E)
{
    int e = blockIdx.x * blockDim.x + threadIdx.x;
    if (e >= E) return;
    atomicAdd(&cnt[erow[e]], 1);
}

__global__ __launch_bounds__(1024)
void scan_kernel(const int* __restrict__ cnt, int* __restrict__ off)
{
    __shared__ int part[1024];
    const int t = threadIdx.x;
    const int per = (NN + 1023) / 1024;   // 10
    int base = t * per;
    int loc[per];
    int s = 0;
    for (int i = 0; i < per; ++i) {
        int idx = base + i;
        int c = (idx < NN) ? cnt[idx] : 0;
        loc[i] = s;                        // exclusive local prefix
        s += c;
    }
    part[t] = s;
    __syncthreads();
    for (int d = 1; d < 1024; d <<= 1) {   // Hillis-Steele over partials
        int v = (t >= d) ? part[t - d] : 0;
        __syncthreads();
        part[t] += v;
        __syncthreads();
    }
    int offset = (t == 0) ? 0 : part[t - 1];
    for (int i = 0; i < per; ++i) {
        int idx = base + i;
        if (idx < NN) off[idx] = offset + loc[i];
    }
}

__global__ void csr_scatter_kernel(const int* __restrict__ erow, const int* __restrict__ ecol,
                                   const float* __restrict__ eval,
                                   const int* __restrict__ off, int* __restrict__ cursor,
                                   int* __restrict__ ecol2, float* __restrict__ eval2, int E)
{
    int e = blockIdx.x * blockDim.x + threadIdx.x;
    if (e >= E) return;
    int r = erow[e];
    int pos = off[r] + atomicAdd(&cursor[r], 1);
    ecol2[pos] = ecol[e];
    eval2[pos] = eval[e];
}

// ---------------------------------------------------------------------------
// 4) gather SPMM: x1[n] = sum_j val_j * x0[col_j]  (CSR, no atomics, f16 out)
//    one thread per (node, 8-half chunk); K = FF*BB = 1056 halves per row
// ---------------------------------------------------------------------------
__global__ void spmm_gather_kernel(const int* __restrict__ off, const int* __restrict__ cnt,
                                   const int* __restrict__ ecol2, const float* __restrict__ eval2,
                                   const _Float16* __restrict__ x0, _Float16* __restrict__ x1)
{
    const int K = FF * BB;            // 1056
    const int PE = K / 8;             // 132 chunks per node
    int idx = blockIdx.x * blockDim.x + threadIdx.x;
    if (idx >= NN * PE) return;
    int n  = idx / PE;
    int k8 = (idx % PE) * 8;
    int j0 = off[n];
    int j1 = j0 + cnt[n];
    float acc[8] = {0.f, 0.f, 0.f, 0.f, 0.f, 0.f, 0.f, 0.f};
    for (int j = j0; j < j1; ++j) {
        float v = eval2[j];
        const _Float16* src = x0 + (size_t)ecol2[j] * K + k8;
        v8h xs = *(const v8h*)src;    // global_load_b128 (L2-resident x0)
#pragma unroll
        for (int i = 0; i < 8; ++i) acc[i] += v * (float)xs[i];
    }
    v8h o;
#pragma unroll
    for (int i = 0; i < 8; ++i) o[i] = (_Float16)acc[i];
    *(v8h*)(x1 + (size_t)n * K + k8) = o;  // global_store_b128
}

// ---------------------------------------------------------------------------
// 5) WMMA GEMM: per block a 16-row tile of A (B*N rows, k = 2f+m interleave),
//    NTILES waves each own one 16-col N-tile. 5 chained v_wmma_f32_16x16x32_f16.
//    GATES epilogue: sigmoid -> r*hx into x0b, u into u_buf.
//    CAND  epilogue: tanh -> out = u*h + (1-u)*c.
// ---------------------------------------------------------------------------
template <int NTILES, bool GATES>
__global__ __launch_bounds__(NTILES * 32)
void gemm_wmma_kernel(const _Float16* __restrict__ x0, const _Float16* __restrict__ x1,
                      const _Float16* __restrict__ Wt,    // [NTILES*16][KPAD] f16
                      const float* __restrict__ bias,
                      const float* __restrict__ hx,
                      _Float16* __restrict__ u_buf,       // [B*N][UU]
                      _Float16* __restrict__ x0b,         // gates only
                      float* __restrict__ out)            // cand only
{
    __shared__ __align__(32) _Float16 As[16][KPAD];
    const int tile = blockIdx.x;
    const int tid = threadIdx.x;

    // cooperative A-tile stage: interleaved k = 2f+m (m=0 -> x0, m=1 -> x1)
    for (int idx = tid; idx < 16 * KPAD; idx += NTILES * 32) {
        int r = idx / KPAD, k = idx % KPAD;
        _Float16 v = (_Float16)0.0f;
        if (k < FF * 2) {
            int f = k >> 1, m = k & 1;
            int bn = tile * 16 + r;
            int b = bn / NN, n = bn % NN;
            size_t o = ((size_t)n * FF + f) * BB + b;
            v = (m == 0) ? x0[o] : x1[o];
        }
        As[r][k] = v;
    }
    __syncthreads();

    const int wave = tid >> 5;          // N-tile index
    const int lane = tid & 31;
    const int hs   = lane >> 4;         // half-wave select
    const int lcol = lane & 15;

    v8f acc = {};
#pragma unroll
    for (int s = 0; s < KS; ++s) {
        const int k0 = s * 32;
        // A fragment (ISA 16-bit 16x32 layout): row = lane&15
        const _Float16* ab = &As[lcol][k0 + 8 * hs];
        v8h alo = *(const v8h*)ab;            // ds_load_b128
        v8h ahi = *(const v8h*)(ab + 16);     // ds_load_b128
        v16h a;
#pragma unroll
        for (int i = 0; i < 8; ++i) { a[i] = alo[i]; a[8 + i] = ahi[i]; }
        // B fragment: col = wave*16 + lcol, K halves k0+8*hs..+7 and k0+16+8*hs..+7
        const _Float16* wb = Wt + ((size_t)(wave * 16 + lcol)) * KPAD + k0 + 8 * hs;
        v8h blo = *(const v8h*)wb;
        v8h bhi = *(const v8h*)(wb + 16);
        v16h bf;
#pragma unroll
        for (int i = 0; i < 8; ++i) { bf[i] = blo[i]; bf[8 + i] = bhi[i]; }
        acc = __builtin_amdgcn_wmma_f32_16x16x32_f16(
            /*neg_a=*/false, a, /*neg_b=*/false, bf,
            /*c_mod=*/(short)0, acc, /*reuse_a=*/false, /*reuse_b=*/false);
    }

    // epilogue: D element v -> row (v + 8*hs), col = wave*16 + lcol
    const int jcol = wave * 16 + lcol;
#pragma unroll
    for (int v = 0; v < 8; ++v) {
        int r = v + 8 * hs;
        int bn = tile * 16 + r;
        int b = bn / NN, n = bn % NN;
        float val = acc[v] + bias[jcol];
        if (GATES) {
            float s = fast_sigmoid(val);
            if (jcol < UU) {                   // reset gate -> state for pass 2
                float h = hx[(size_t)b * NN * UU + (size_t)n * UU + jcol];
                x0b[((size_t)n * FF + (IND + jcol)) * BB + b] = (_Float16)(s * h);
            } else {                           // update gate
                u_buf[(size_t)bn * UU + (jcol - UU)] = (_Float16)s;
            }
        } else {
            float c = fast_tanh(val);
            float u = (float)u_buf[(size_t)bn * UU + jcol];
            float h = hx[(size_t)b * NN * UU + (size_t)n * UU + jcol];
            out[(size_t)b * NN * UU + (size_t)n * UU + jcol] = u * h + (1.f - u) * c;
        }
    }
}

// ---------------------------------------------------------------------------
extern "C" void kernel_launch(void* const* d_in, const int* in_sizes, int n_in,
                              void* d_out, int out_size, void* d_ws, size_t ws_size,
                              hipStream_t stream)
{
    const float* inputs = (const float*)d_in[0];
    const float* hx     = (const float*)d_in[1];
    const float* eval   = (const float*)d_in[2];
    const float* w1_mu  = (const float*)d_in[3];
    const float* w1_ls  = (const float*)d_in[4];
    const float* b1_mu  = (const float*)d_in[5];
    const float* b1_ls  = (const float*)d_in[6];
    const float* w2_mu  = (const float*)d_in[7];
    const float* w2_ls  = (const float*)d_in[8];
    const float* b2_mu  = (const float*)d_in[9];
    const float* b2_ls  = (const float*)d_in[10];
    const float* e_w1   = (const float*)d_in[11];
    const float* e_b1   = (const float*)d_in[12];
    const float* e_w2   = (const float*)d_in[13];
    const float* e_b2   = (const float*)d_in[14];
    const int*   erow   = (const int*)d_in[15];
    const int*   ecol   = (const int*)d_in[16];
    float* out = (float*)d_out;
    const int E = in_sizes[15];

    // workspace carve-up (256B aligned slices)
    char* p = (char*)d_ws;
    auto carve = [&](size_t bytes) -> char* {
        char* r = p; p += (bytes + 255) & ~(size_t)255; return r;
    };
    _Float16* W1t  = (_Float16*)carve((size_t)2 * UU * KPAD * 2);
    _Float16* W2t  = (_Float16*)carve((size_t)UU * KPAD * 2);
    float*    B1f  = (float*)carve(2 * UU * 4);
    float*    B2f  = (float*)carve(UU * 4);
    const size_t xbytes = (size_t)NN * FF * BB * 2;    // f16 feature planes
    _Float16* x0a  = (_Float16*)carve(xbytes);
    _Float16* x0b  = (_Float16*)carve(xbytes);
    _Float16* x1a  = (_Float16*)carve(xbytes);
    _Float16* x1b  = (_Float16*)carve(xbytes);
    _Float16* ubuf = (_Float16*)carve((size_t)BB * NN * UU * 2);
    int*   rcnt    = (int*)carve((size_t)NN * 4);
    int*   roff    = (int*)carve((size_t)NN * 4);
    int*   rcur    = (int*)carve((size_t)NN * 4);
    int*   ecol2   = (int*)carve((size_t)E * 4);
    float* eval2   = (float*)carve((size_t)E * 4);

    // --- CSR build (cheap; rebuilt every call for determinism of work) ---
    hipMemsetAsync(rcnt, 0, (size_t)NN * 4, stream);
    hipMemsetAsync(rcur, 0, (size_t)NN * 4, stream);
    csr_count_kernel<<<(E + 255) / 256, 256, 0, stream>>>(erow, rcnt, E);
    scan_kernel<<<1, 1024, 0, stream>>>(rcnt, roff);
    csr_scatter_kernel<<<(E + 255) / 256, 256, 0, stream>>>(
        erow, ecol, eval, roff, rcur, ecol2, eval2, E);

    // --- weights + input staging ---
    {
        int total = 3 * UU * KPAD + 3 * UU;
        reparam_kernel<<<(total + 255) / 256, 256, 0, stream>>>(
            w1_mu, w1_ls, e_w1, b1_mu, b1_ls, e_b1,
            w2_mu, w2_ls, e_w2, b2_mu, b2_ls, e_b2, W1t, W2t, B1f, B2f);
    }
    {
        int total = NN * FF * BB;
        assemble_kernel<<<(total + 255) / 256, 256, 0, stream>>>(inputs, hx, x0a, x0b);
    }

    const int spmm_total = NN * (FF * BB / 8);   // 1.32M threads
    // --- pass 1: diffusion + gates ---
    spmm_gather_kernel<<<(spmm_total + 255) / 256, 256, 0, stream>>>(
        roff, rcnt, ecol2, eval2, x0a, x1a);
    gemm_wmma_kernel<8, true><<<(BB * NN) / 16, 256, 0, stream>>>(
        x0a, x1a, W1t, B1f, hx, ubuf, x0b, nullptr);
    // --- pass 2: diffusion on (inputs, r*hx) + candidate + blend ---
    spmm_gather_kernel<<<(spmm_total + 255) / 256, 256, 0, stream>>>(
        roff, rcnt, ecol2, eval2, x0b, x1b);
    gemm_wmma_kernel<4, false><<<(BB * NN) / 16, 128, 0, stream>>>(
        x0b, x1b, W2t, B2f, hx, ubuf, nullptr, out);
}
